// Attention_72275709657473
// MI455X (gfx1250) — matrix operand order repaired and checked by reading, not compiled, not run on hardware
//
#include <hip/hip_runtime.h>
#include <hip/hip_bf16.h>

// ---------------------------------------------------------------------------
// Causal MHA for MI455X (gfx1250, wave32, WMMA).
//   B=4, S=2048, E=1024, H=16, D=64.  All matmuls use v_wmma_f32_16x16x32_f16.
// Pipeline:
//   0) cvt_w     : Wq/Wk/Wv/Wo fp32 -> f16 (weights reused ~64x in GEMMs)
//   1) qkv_gemm  : 64x64 wave tiles; Q,K -> [b,h,s,d] f16; V -> [b,h,d,s] f16
//   2) attn      : flash attention, 1 wave per (b,h, 16-row q tile), 32 keys/step
//   3) out_gemm  : f16 attn-out @ Wo^T + bo -> fp32 [b,s,e]
// ---------------------------------------------------------------------------

typedef __attribute__((ext_vector_type(16))) _Float16 v16h;
typedef __attribute__((ext_vector_type(8)))  _Float16 v8h;
typedef __attribute__((ext_vector_type(8)))  float    v8f;
typedef __attribute__((ext_vector_type(4)))  float    v4f;

#define EMBED 1024
#define NHEAD 16
#define HDIM  64
#define SEQ   2048
#define BATCH 4
#define MROWS (BATCH * SEQ)          // 8192
#define ATT_SCALE 0.125f             // 64^-0.5

__device__ __forceinline__ v8f wmma16(v16h a, v16h b, v8f c) {
  return __builtin_amdgcn_wmma_f32_16x16x32_f16(
      /*neg_a=*/false, a, /*neg_b=*/false, b,
      /*c_mod=*/(short)0, c, /*reuse_a=*/false, /*reuse_b=*/false);
}

// Pack two 8-float chunks into one 16-element f16 fragment (with conversion).
__device__ __forceinline__ v16h cvt_pack(v8f lo, v8f hi) {
  v16h r;
#pragma unroll
  for (int i = 0; i < 8; ++i) { r[i] = (_Float16)lo[i]; r[i + 8] = (_Float16)hi[i]; }
  return r;
}
__device__ __forceinline__ v16h pack16(v8h lo, v8h hi) {
  v16h r;
#pragma unroll
  for (int i = 0; i < 8; ++i) { r[i] = lo[i]; r[i + 8] = hi[i]; }
  return r;
}

// ---------------------------------------------------------------------------
// Kernel 0: weight conversion fp32 -> f16.  grid = (512, 4), block = 256.
// Each thread converts 8 contiguous elements (two b128 loads -> one b128 store).
// ---------------------------------------------------------------------------
__global__ __launch_bounds__(256) void mha_cvt_w(
    const float* __restrict__ W0, const float* __restrict__ W1,
    const float* __restrict__ W2, const float* __restrict__ W3,
    _Float16* __restrict__ O0, _Float16* __restrict__ O1,
    _Float16* __restrict__ O2, _Float16* __restrict__ O3) {
  const int w = blockIdx.y;
  const float* src = (w == 0) ? W0 : (w == 1) ? W1 : (w == 2) ? W2 : W3;
  _Float16*    dst = (w == 0) ? O0 : (w == 1) ? O1 : (w == 2) ? O2 : O3;
  const size_t i = ((size_t)blockIdx.x * blockDim.x + threadIdx.x) * 8;
  v4f a = *(const v4f*)(src + i);
  v4f b = *(const v4f*)(src + i + 4);
  v8h r;
#pragma unroll
  for (int k = 0; k < 4; ++k) { r[k] = (_Float16)a[k]; r[k + 4] = (_Float16)b[k]; }
  *(v8h*)(dst + i) = r;
}

// ---------------------------------------------------------------------------
// Kernel 1: fused QKV projection.  grid = (8, 64, 3), block = 128 (4 waves).
// Each wave: 64x64 output tile (4x4 WMMA frags, 16 WMMA / K-step).
// A: fp32 activations converted on the fly; B: pre-converted f16 weights.
// blockIdx.z: 0 -> Q, 1 -> K, 2 -> V(transposed store).
// ---------------------------------------------------------------------------
__global__ __launch_bounds__(128) void mha_qkv_gemm(
    const float* __restrict__ inq, const float* __restrict__ ink,
    const float* __restrict__ inv,
    const _Float16* __restrict__ Wq, const _Float16* __restrict__ Wk,
    const _Float16* __restrict__ Wv,
    const float* __restrict__ bq, const float* __restrict__ bk,
    const float* __restrict__ bv,
    _Float16* __restrict__ outq, _Float16* __restrict__ outk,
    _Float16* __restrict__ outvt) {
  const int mode = blockIdx.z;
  const float*    X    = (mode == 0) ? inq : (mode == 1) ? ink : inv;
  const _Float16* W    = (mode == 0) ? Wq  : (mode == 1) ? Wk  : Wv;
  const float*    bias = (mode == 0) ? bq  : (mode == 1) ? bk  : bv;
  _Float16*       O    = (mode == 0) ? outq : (mode == 1) ? outk : outvt;

  const int lane = threadIdx.x & 31;
  const int wave = threadIdx.x >> 5;
  const int ln   = lane & 15;           // A row / B row / D column within tile
  const int hi   = lane >> 4;           // half-wave select
  const int mbase = blockIdx.y * 128 + (wave >> 1) * 64;
  const int nbase = blockIdx.x * 128 + (wave & 1) * 64;

  const float* xrow[4];
  const _Float16* wrow[4];
#pragma unroll
  for (int t = 0; t < 4; ++t) {
    xrow[t] = X + (size_t)(mbase + t * 16 + ln) * EMBED;
    wrow[t] = W + (size_t)(nbase + t * 16 + ln) * EMBED;
  }

  v8f acc[4][4] = {};
  for (int kk = 0; kk < EMBED; kk += 32) {
    v16h a[4], b[4];
#pragma unroll
    for (int t = 0; t < 4; ++t) {
      // A: element i -> k = kk + 16*hi + i
      a[t] = cvt_pack(*(const v8f*)(xrow[t] + kk + 16 * hi),
                      *(const v8f*)(xrow[t] + kk + 16 * hi + 8));
      // B: element i -> k = kk + 16*hi + i  (16 contiguous f16, 32B aligned)
      b[t] = *(const v16h*)(wrow[t] + kk + 16 * hi);
    }
#pragma unroll
    for (int mi = 0; mi < 4; ++mi)
#pragma unroll
      for (int ni = 0; ni < 4; ++ni)
        acc[mi][ni] = wmma16(a[mi], b[ni], acc[mi][ni]);
  }

#pragma unroll
  for (int mi = 0; mi < 4; ++mi) {
#pragma unroll
    for (int ni = 0; ni < 4; ++ni) {
      const int nfeat = nbase + ni * 16 + ln;   // output feature for this lane
      const float bv_ = bias[nfeat];
      const int h = nfeat >> 6, d = nfeat & 63;
#pragma unroll
      for (int r = 0; r < 8; ++r) {             // D layout: m = r + 8*hi
        const int m = mbase + mi * 16 + r + 8 * hi;
        const int bidx = m >> 11, s = m & (SEQ - 1);
        const float val = acc[mi][ni][r] + bv_;
        if (mode < 2)   // Q,K: [b,h,s,d]
          O[(((size_t)(bidx * NHEAD + h)) * SEQ + s) * HDIM + d] = (_Float16)val;
        else            // V: transposed [b,h,d,s]
          O[(((size_t)(bidx * NHEAD + h)) * HDIM + d) * SEQ + s] = (_Float16)val;
      }
    }
  }
}

// ---------------------------------------------------------------------------
// Kernel 2: causal flash attention.  grid = 8192 blocks x 32 threads (1 wave).
// block -> (bh, qtile): 16 query rows, loop over 32-key blocks.
// ---------------------------------------------------------------------------
__global__ __launch_bounds__(32) void mha_attn(
    const _Float16* __restrict__ Q, const _Float16* __restrict__ K,
    const _Float16* __restrict__ Vt, _Float16* __restrict__ AO) {
  __shared__ _Float16 Plds[16][32];   // P tile staging: C/D layout -> A layout

  const int lane = threadIdx.x & 31;
  const int ln = lane & 15, hi = lane >> 4;
  const int tile  = blockIdx.x;
  const int qt    = tile & (SEQ / 16 - 1);     // 0..127
  const int bh    = tile >> 7;                 // b*16 + h
  const int qbase = qt * 16;

  const _Float16* Kbh = K  + (size_t)bh * SEQ * HDIM;
  const _Float16* Vbh = Vt + (size_t)bh * HDIM * SEQ;

  // Q A-fragments (row = ln), d split into two K=32 chunks
  const _Float16* Qp = Q + ((size_t)bh * SEQ + qbase + ln) * HDIM;
  v16h qa[2];
#pragma unroll
  for (int c = 0; c < 2; ++c)
    qa[c] = pack16(*(const v8h*)(Qp + 32 * c + 16 * hi),
                   *(const v8h*)(Qp + 32 * c + 16 * hi + 8));

  v8f acc[4] = {};                 // O accum: 16 rows x 64 d (4 n-chunks)
  float mrow[8], lrow[8];
#pragma unroll
  for (int r = 0; r < 8; ++r) { mrow[r] = -INFINITY; lrow[r] = 0.0f; }

  const int nblocks = (qbase + 16 + 31) >> 5;  // causal: keys < qbase+16
  for (int j = 0; j < nblocks; ++j) {
    const int jbase = j * 32;
    // ---- S = Q * K^T : two 16x16 score tiles (keys jbase.., jbase+16..)
    v8f s0 = {}, s1 = {};
#pragma unroll
    for (int c = 0; c < 2; ++c) {   // d chunks of 32
      v16h kb0 = *(const v16h*)(Kbh + (size_t)(jbase + ln) * HDIM + 32 * c + 16 * hi);
      v16h kb1 = *(const v16h*)(Kbh + (size_t)(jbase + 16 + ln) * HDIM + 32 * c + 16 * hi);
      s0 = wmma16(qa[c], kb0, s0);
      s1 = wmma16(qa[c], kb1, s1);
    }
    // ---- online softmax (rows m = r+8*hi live across the 16-lane half group)
    const bool need_mask = (jbase + 32 > qbase);
#pragma unroll
    for (int r = 0; r < 8; ++r) {
      const int qg = qbase + r + 8 * hi;
      float x0 = s0[r] * ATT_SCALE;
      float x1 = s1[r] * ATT_SCALE;
      if (need_mask) {
        if (jbase + ln > qg)      x0 = -INFINITY;
        if (jbase + 16 + ln > qg) x1 = -INFINITY;
      }
      float t = fmaxf(x0, x1);
      t = fmaxf(t, __shfl_xor(t, 1, 32));
      t = fmaxf(t, __shfl_xor(t, 2, 32));
      t = fmaxf(t, __shfl_xor(t, 4, 32));
      t = fmaxf(t, __shfl_xor(t, 8, 32));
      const float newm = fmaxf(mrow[r], t);
      const float alpha = __expf(mrow[r] - newm);   // exp(-inf)=0 first time
      const float p0 = __expf(x0 - newm);
      const float p1 = __expf(x1 - newm);
      float rs = p0 + p1;
      rs += __shfl_xor(rs, 1, 32);
      rs += __shfl_xor(rs, 2, 32);
      rs += __shfl_xor(rs, 4, 32);
      rs += __shfl_xor(rs, 8, 32);
      lrow[r] = lrow[r] * alpha + rs;
      mrow[r] = newm;
#pragma unroll
      for (int c = 0; c < 4; ++c) acc[c][r] *= alpha;
      Plds[r + 8 * hi][ln]      = (_Float16)p0;
      Plds[r + 8 * hi][ln + 16] = (_Float16)p1;
    }
    __syncthreads();
    // ---- P (C/D layout) -> A fragment layout
    v16h pa;
#pragma unroll
    for (int i = 0; i < 8; ++i) {
      pa[i]     = Plds[ln][16 * hi + i];
      pa[i + 8] = Plds[ln][16 * hi + 8 + i];
    }
    __syncthreads();
    // ---- O += P * V   (V transposed => contiguous B-fragment loads)
#pragma unroll
    for (int c = 0; c < 4; ++c) {
      v16h vb = *(const v16h*)(Vbh + (size_t)(16 * c + ln) * SEQ + jbase + 16 * hi);
      acc[c] = wmma16(pa, vb, acc[c]);
    }
  }

  // ---- normalize and emit row-major [b, s, e] f16
  const int b = bh >> 4, h = bh & 15;
#pragma unroll
  for (int r = 0; r < 8; ++r) {
    const float inv = 1.0f / lrow[r];
    const int sg = qbase + r + 8 * hi;
#pragma unroll
    for (int c = 0; c < 4; ++c) {
      const int d = 16 * c + ln;
      AO[((size_t)(b * SEQ + sg)) * EMBED + h * HDIM + d] =
          (_Float16)(acc[c][r] * inv);
    }
  }
}

// ---------------------------------------------------------------------------
// Kernel 3: output projection (f16 A, f16 B, fp32 result).
// grid = (8, 64), block = 128; 64x64 per wave.
// ---------------------------------------------------------------------------
__global__ __launch_bounds__(128) void mha_out_gemm(
    const _Float16* __restrict__ A, const _Float16* __restrict__ Wo,
    const float* __restrict__ bo, float* __restrict__ out) {
  const int lane = threadIdx.x & 31;
  const int wave = threadIdx.x >> 5;
  const int ln = lane & 15, hi = lane >> 4;
  const int mbase = blockIdx.y * 128 + (wave >> 1) * 64;
  const int nbase = blockIdx.x * 128 + (wave & 1) * 64;

  const _Float16* arow[4];
  const _Float16* wrow[4];
#pragma unroll
  for (int t = 0; t < 4; ++t) {
    arow[t] = A  + (size_t)(mbase + t * 16 + ln) * EMBED;
    wrow[t] = Wo + (size_t)(nbase + t * 16 + ln) * EMBED;
  }

  v8f acc[4][4] = {};
  for (int kk = 0; kk < EMBED; kk += 32) {
    v16h a[4], b[4];
#pragma unroll
    for (int t = 0; t < 4; ++t) {
      a[t] = pack16(*(const v8h*)(arow[t] + kk + 16 * hi),
                    *(const v8h*)(arow[t] + kk + 16 * hi + 8));
      b[t] = *(const v16h*)(wrow[t] + kk + 16 * hi);
    }
#pragma unroll
    for (int mi = 0; mi < 4; ++mi)
#pragma unroll
      for (int ni = 0; ni < 4; ++ni)
        acc[mi][ni] = wmma16(a[mi], b[ni], acc[mi][ni]);
  }

#pragma unroll
  for (int mi = 0; mi < 4; ++mi) {
#pragma unroll
    for (int ni = 0; ni < 4; ++ni) {
      const int nfeat = nbase + ni * 16 + ln;
      const float bias = bo[nfeat];
#pragma unroll
      for (int r = 0; r < 8; ++r) {
        const int m = mbase + mi * 16 + r + 8 * hi;
        out[(size_t)m * EMBED + nfeat] = acc[mi][ni][r] + bias;
      }
    }
  }
}

// ---------------------------------------------------------------------------
extern "C" void kernel_launch(void* const* d_in, const int* in_sizes, int n_in,
                              void* d_out, int out_size, void* d_ws, size_t ws_size,
                              hipStream_t stream) {
  const float* query = (const float*)d_in[0];
  const float* key_  = (const float*)d_in[1];
  const float* value = (const float*)d_in[2];
  const float* Wq = (const float*)d_in[3];
  const float* bq = (const float*)d_in[4];
  const float* Wk = (const float*)d_in[5];
  const float* bk = (const float*)d_in[6];
  const float* Wv = (const float*)d_in[7];
  const float* bv = (const float*)d_in[8];
  const float* Wo = (const float*)d_in[9];
  const float* bo = (const float*)d_in[10];

  const size_t NW = (size_t)EMBED * EMBED;                // 1,048,576 elems
  const size_t NQ = (size_t)BATCH * NHEAD * SEQ * HDIM;   // 8,388,608 elems
  _Float16* wq16 = (_Float16*)d_ws;
  _Float16* wk16 = wq16 + NW;
  _Float16* wv16 = wk16 + NW;
  _Float16* wo16 = wv16 + NW;
  _Float16* qb   = wo16 + NW;               // [b,h,s,d]
  _Float16* kb   = qb + NQ;                 // [b,h,s,d]
  _Float16* vtb  = kb + NQ;                 // [b,h,d,s]
  _Float16* aob  = vtb + NQ;                // [b,s,e]  (total ws ~= 75 MB)

  dim3 gw(NW / (256 * 8), 4);               // (512, 4)
  mha_cvt_w<<<gw, 256, 0, stream>>>(Wq, Wk, Wv, Wo, wq16, wk16, wv16, wo16);

  dim3 gp(EMBED / 128, MROWS / 128, 3);     // (8, 64, 3)
  mha_qkv_gemm<<<gp, 128, 0, stream>>>(query, key_, value, wq16, wk16, wv16,
                                       bq, bk, bv, qb, kb, vtb);

  mha_attn<<<BATCH * NHEAD * (SEQ / 16), 32, 0, stream>>>(qb, kb, vtb, aob);

  dim3 go(EMBED / 128, MROWS / 128);        // (8, 64)
  mha_out_gemm<<<go, 128, 0, stream>>>(aob, wo16, bo, (float*)d_out);
}
